// HierarchicalDecoder_75144747810989
// MI455X (gfx1250) — matrix operand order repaired and checked by reading, not compiled, not run on hardware
//
#include <hip/hip_runtime.h>
#include <hip/hip_bf16.h>

typedef __attribute__((ext_vector_type(16))) _Float16 v16h;
typedef __attribute__((ext_vector_type(8)))  _Float16 v8h;
typedef __attribute__((ext_vector_type(8)))  float    v8f;

#define B_     64
#define LAT_   512
#define HC_    1024
#define HD_    1024
#define NB_    16
#define S_     16
#define E_     128
#define NOTES_ 128
#define OUT_   (NOTES_ * 3)      // 384
#define G_     (4 * HC_)         // 4096 gate width
#define RB_    (NB_ * B_)        // 1024 decoder batch (bars folded into batch)
#define TR_    (S_ * RB_)        // 16384 total decoder rows

// ---------------------------------------------------------------------------
// f32 -> f16 2D convert with destination stride/column offset (weight prep,
// also used to build K-concatenated [wih | whh] weight blocks).
// ---------------------------------------------------------------------------
__global__ void k_cvt2d(const float* __restrict__ src, _Float16* __restrict__ dst,
                        int rows, int cols, int dst_ld, int dst_coloff) {
    int i = blockIdx.x * blockDim.x + threadIdx.x;
    int total = rows * cols;
    if (i >= total) return;
    int r = i / cols, c = i - r * cols;
    dst[(long long)r * dst_ld + dst_coloff + c] = (_Float16)src[i];
}

// ---------------------------------------------------------------------------
// WMMA GEMM:  out[M,N] = A[M,K](f16) * W[N,K](f16)^T  (+bias0 +bias1 +Cin)
// One wave32 per 16(M) x 64(N) tile, K in chunks of 32 via v_wmma_f32_16x16x32_f16.
// Fragment layouts (A and B share a contiguous k order; C/D is the hardware layout):
//   A (16x32): lane m=lane&15, hi=lane>>4; elems 0..15 <- K = k0+hi*16+e
//              (one 32B load)
//   B (32x16): lane n=lane&15, hi=lane>>4; elems 0..15 <- K = k0+hi*16+e
//              (one 32B load; W row-major [N,K] makes this contiguous)
//   C/D (16x16 f32): VGPR v -> M = hi*8+v, N = lane&15
// Requires M%16==0, N%64==0, K%32==0 (all call sites satisfy this).
// ---------------------------------------------------------------------------
__global__ __launch_bounds__(32) void k_wmma_gemm(
    const _Float16* __restrict__ A, int lda,
    const _Float16* __restrict__ W, int ldw,
    float* __restrict__ C, int ldc,
    _Float16* __restrict__ C16, int ldc16,
    const float* __restrict__ bias0, const float* __restrict__ bias1,
    const float* __restrict__ Cin, int ldcin,
    int K)
{
    const int lane = threadIdx.x & 31;
    const int m16  = lane & 15;
    const int hi   = lane >> 4;
    const int col0 = blockIdx.x * 64;
    const int row0 = blockIdx.y * 16;

    v8f acc0 = {0.f, 0.f, 0.f, 0.f, 0.f, 0.f, 0.f, 0.f};
    v8f acc1 = acc0, acc2 = acc0, acc3 = acc0;

    const _Float16* Arow = A + (long long)(row0 + m16) * lda;
    const _Float16* W0   = W + (long long)(col0 +  0 + m16) * ldw;
    const _Float16* W1   = W + (long long)(col0 + 16 + m16) * ldw;
    const _Float16* W2   = W + (long long)(col0 + 32 + m16) * ldw;
    const _Float16* W3   = W + (long long)(col0 + 48 + m16) * ldw;

    for (int k0 = 0; k0 < K; k0 += 32) {
        if (k0 + 32 < K) {
            __builtin_prefetch(Arow + k0 + 32, 0, 1);   // global_prefetch_b8
            __builtin_prefetch(W0 + k0 + 32, 0, 1);
        }
        v16h a = *(const v16h*)(Arow + k0 + hi * 16);
        v16h b0 = *(const v16h*)(W0 + k0 + hi * 16);
        v16h b1 = *(const v16h*)(W1 + k0 + hi * 16);
        v16h b2 = *(const v16h*)(W2 + k0 + hi * 16);
        v16h b3 = *(const v16h*)(W3 + k0 + hi * 16);
        acc0 = __builtin_amdgcn_wmma_f32_16x16x32_f16(false, a, false, b0, (short)0, acc0, false, false);
        acc1 = __builtin_amdgcn_wmma_f32_16x16x32_f16(false, a, false, b1, (short)0, acc1, false, false);
        acc2 = __builtin_amdgcn_wmma_f32_16x16x32_f16(false, a, false, b2, (short)0, acc2, false, false);
        acc3 = __builtin_amdgcn_wmma_f32_16x16x32_f16(false, a, false, b3, (short)0, acc3, false, false);
    }

    v8f accs[4] = {acc0, acc1, acc2, acc3};
#pragma unroll
    for (int t = 0; t < 4; ++t) {
        const int col = col0 + t * 16 + m16;
        float badd = 0.f;
        if (bias0) badd += bias0[col];
        if (bias1) badd += bias1[col];
#pragma unroll
        for (int v = 0; v < 8; ++v) {
            const int row = row0 + hi * 8 + v;
            float val = accs[t][v] + badd;
            if (Cin) val += Cin[(long long)row * ldcin + col];
            if (C)   C[(long long)row * ldc + col] = val;
            if (C16) C16[(long long)row * ldc16 + col] = (_Float16)val;
        }
    }
}

// ---------------------------------------------------------------------------
// LSTM elementwise: gates[M,4H] (i,f,g,o blocks) + c_state -> new c (f32, in
// place) and h written as f16 into up to two strided destinations (the
// [h0|h1] concat buffer for the next GEMMs, plus optional collector buffer).
// ---------------------------------------------------------------------------
__global__ void k_lstm_cell(const float* __restrict__ gates,
                            float* __restrict__ c_state,
                            _Float16* __restrict__ hA, int ldA,
                            _Float16* __restrict__ hB, int ldB,
                            int M, int H)
{
    int i = blockIdx.x * blockDim.x + threadIdx.x;
    if (i >= M * H) return;
    int r = i / H, j = i - r * H;
    const float* g = gates + (long long)r * 4 * H;
    float ig = 1.f / (1.f + __expf(-g[j]));
    float fg = 1.f / (1.f + __expf(-g[H + j]));
    float gg = tanhf(g[2 * H + j]);
    float og = 1.f / (1.f + __expf(-g[3 * H + j]));
    float c  = fg * c_state[i] + ig * gg;
    c_state[i] = c;
    float h = og * tanhf(c);
    hA[(long long)r * ldA + j] = (_Float16)h;
    if (hB) hB[(long long)r * ldB + j] = (_Float16)h;
}

// ---------------------------------------------------------------------------
// Scatter [M,4H] init GEMM output (cols: h_l0, h_l1, c_l0, c_l1) into the
// f16 [h0|h1] concat buffer and f32 c-state buffers.
// ---------------------------------------------------------------------------
__global__ void k_scatter_init(const float* __restrict__ ini, _Float16* __restrict__ cat,
                               float* __restrict__ c0, float* __restrict__ c1, int M, int H)
{
    int i = blockIdx.x * blockDim.x + threadIdx.x;
    if (i >= M * H) return;
    int r = i / H, j = i - r * H;
    const float* row = ini + (long long)r * 4 * H;
    cat[(long long)r * 2 * H + j]     = (_Float16)row[j];
    cat[(long long)r * 2 * H + H + j] = (_Float16)row[H + j];
    c0[i] = row[2 * H + j];
    c1[i] = row[3 * H + j];
}

// ---------------------------------------------------------------------------
// Teacher-forced prev input: prev[s, nb*B+b, note] = (nb*16+s==0) ? 0
//   : target[b, nb*16+s-1, note], stored as f16 GEMM operand.
// ---------------------------------------------------------------------------
__global__ void k_build_prev(const int* __restrict__ target, _Float16* __restrict__ prev)
{
    int i = blockIdx.x * blockDim.x + threadIdx.x;
    if (i >= S_ * RB_ * NOTES_) return;
    int note = i & 127;
    int r    = (i >> 7) & 1023;       // nb*64 + b
    int s    = i >> 17;
    int nb   = r >> 6;
    int b    = r & 63;
    int gt   = nb * S_ + s;
    float v  = 0.f;
    if (gt > 0) v = (float)target[((long long)b * (NB_ * S_) + (gt - 1)) * NOTES_ + note];
    prev[i] = (_Float16)v;
}

// Broadcast conductor bar embedding into xs columns [E_, E_+HC_) for every step.
__global__ void k_bcast_bar(const _Float16* __restrict__ condh, _Float16* __restrict__ xs)
{
    int i = blockIdx.x * blockDim.x + threadIdx.x;
    if (i >= S_ * RB_ * HC_) return;
    int j = i & 1023;
    int r = (i >> 10) & 1023;
    int s = i >> 20;
    xs[((long long)s * RB_ + r) * (E_ + HC_) + E_ + j] = condh[(long long)r * HC_ + j];
}

// logits[s*1024 + nb*64 + b, o] -> out[b, nb*16 + s, o]  (o = note*3 + state)
__global__ void k_scatter_out(const float* __restrict__ logits, float* __restrict__ out)
{
    int i = blockIdx.x * blockDim.x + threadIdx.x;
    if (i >= TR_ * OUT_) return;
    int o   = i % OUT_;
    int row = i / OUT_;
    int b   = row & 63;
    int nb  = (row >> 6) & 15;
    int s   = row >> 10;
    out[(((long long)b * (NB_ * S_) + nb * S_ + s) * OUT_) + o] = logits[i];
}

// ---------------------------------------------------------------------------
extern "C" void kernel_launch(void* const* d_in, const int* in_sizes, int n_in,
                              void* d_out, int out_size, void* d_ws, size_t ws_size,
                              hipStream_t stream)
{
    (void)in_sizes; (void)n_in; (void)out_size; (void)ws_size;

    const float* z      = (const float*)d_in[0];
    const int*   tgt    = (const int*)d_in[1];
    const float* w_zc   = (const float*)d_in[2];
    const float* b_zc   = (const float*)d_in[3];
    /* d_in[4] = c_wih0: unused — conductor layer-0 input is all zeros */
    const float* c_whh0 = (const float*)d_in[5];
    const float* c_bih0 = (const float*)d_in[6];
    const float* c_bhh0 = (const float*)d_in[7];
    const float* c_wih1 = (const float*)d_in[8];
    const float* c_whh1 = (const float*)d_in[9];
    const float* c_bih1 = (const float*)d_in[10];
    const float* c_bhh1 = (const float*)d_in[11];
    const float* np_w   = (const float*)d_in[12];
    const float* np_b   = (const float*)d_in[13];
    const float* cd_w   = (const float*)d_in[14];
    const float* cd_b   = (const float*)d_in[15];
    const float* d_wih0 = (const float*)d_in[16];
    const float* d_whh0 = (const float*)d_in[17];
    const float* d_bih0 = (const float*)d_in[18];
    const float* d_bhh0 = (const float*)d_in[19];
    const float* d_wih1 = (const float*)d_in[20];
    const float* d_whh1 = (const float*)d_in[21];
    const float* d_bih1 = (const float*)d_in[22];
    const float* d_bhh1 = (const float*)d_in[23];
    const float* op_w   = (const float*)d_in[24];
    const float* op_b   = (const float*)d_in[25];
    float* out = (float*)d_out;

    // ---- workspace carving (256B aligned) ----
    char* p = (char*)d_ws;
    auto carve = [&](size_t bytes) -> void* {
        void* r = (void*)p;
        p += (bytes + 255) & ~(size_t)255;
        return r;
    };
    _Float16* Wzch  = (_Float16*)carve((size_t)G_ * LAT_ * 2);
    _Float16* Wc0h  = (_Float16*)carve((size_t)G_ * HC_ * 2);
    _Float16* Wc1c  = (_Float16*)carve((size_t)G_ * 2 * HC_ * 2);   // [c_wih1 | c_whh1]
    _Float16* Wcdh  = (_Float16*)carve((size_t)G_ * HC_ * 2);
    _Float16* Wd0i  = (_Float16*)carve((size_t)G_ * (E_ + HC_) * 2);
    _Float16* Wd0h  = (_Float16*)carve((size_t)G_ * HD_ * 2);
    _Float16* Wd1c  = (_Float16*)carve((size_t)G_ * 2 * HD_ * 2);   // [d_wih1 | d_whh1]
    _Float16* Wnph  = (_Float16*)carve((size_t)E_ * NOTES_ * 2);
    _Float16* Woph  = (_Float16*)carve((size_t)OUT_ * HD_ * 2);
    _Float16* zh    = (_Float16*)carve((size_t)B_ * LAT_ * 2);
    float*    inibuf= (float*)carve((size_t)B_ * G_ * 4);
    float*    gat_c = (float*)carve((size_t)B_ * G_ * 4);
    _Float16* cat_c = (_Float16*)carve((size_t)B_ * 2 * HC_ * 2);   // [h0 | h1] conductor
    float*    cc0   = (float*)carve((size_t)B_ * HC_ * 4);
    float*    cc1   = (float*)carve((size_t)B_ * HC_ * 4);
    _Float16* condh = (_Float16*)carve((size_t)RB_ * HC_ * 2);      // per-bar top h
    float*    gat_d = (float*)carve((size_t)RB_ * G_ * 4);          // also holds di
    _Float16* cat_d = (_Float16*)carve((size_t)RB_ * 2 * HD_ * 2);  // [h0 | h1] decoder
    float*    dc0   = (float*)carve((size_t)RB_ * HD_ * 4);
    float*    dc1   = (float*)carve((size_t)RB_ * HD_ * 4);
    _Float16* prevb = (_Float16*)carve((size_t)TR_ * NOTES_ * 2);
    _Float16* xs    = (_Float16*)carve((size_t)TR_ * (E_ + HC_) * 2);
    _Float16* ys    = (_Float16*)carve((size_t)TR_ * HD_ * 2);
    float*    lgts  = (float*)carve((size_t)TR_ * OUT_ * 4);

    auto cvt = [&](const float* s, _Float16* d, int rows, int cols, int ld, int off) {
        int tot = rows * cols;
        k_cvt2d<<<(tot + 255) / 256, 256, 0, stream>>>(s, d, rows, cols, ld, off);
    };
    auto gemm = [&](const _Float16* A, int lda, const _Float16* W, int ldw,
                    float* C, int ldc, _Float16* C16, int ldc16,
                    const float* b0, const float* b1,
                    const float* Cin, int ldcin, int M, int N, int K) {
        dim3 grid(N / 64, M / 16);
        k_wmma_gemm<<<grid, 32, 0, stream>>>(A, lda, W, ldw, C, ldc, C16, ldc16,
                                             b0, b1, Cin, ldcin, K);
    };

    // ---- Phase 0: weight / input conversion to f16 ----
    cvt(w_zc,   Wzch, G_, LAT_, LAT_, 0);
    cvt(c_whh0, Wc0h, G_, HC_, HC_, 0);
    cvt(c_wih1, Wc1c, G_, HC_, 2 * HC_, 0);
    cvt(c_whh1, Wc1c, G_, HC_, 2 * HC_, HC_);
    cvt(cd_w,   Wcdh, G_, HC_, HC_, 0);
    cvt(d_wih0, Wd0i, G_, E_ + HC_, E_ + HC_, 0);
    cvt(d_whh0, Wd0h, G_, HD_, HD_, 0);
    cvt(d_wih1, Wd1c, G_, HD_, 2 * HD_, 0);
    cvt(d_whh1, Wd1c, G_, HD_, 2 * HD_, HD_);
    cvt(np_w,   Wnph, E_, NOTES_, NOTES_, 0);
    cvt(op_w,   Woph, OUT_, HD_, HD_, 0);
    cvt(z,      zh,   B_, LAT_, LAT_, 0);

    // ---- Phase 1: conductor initial state: ini = z @ w_zc^T + b_zc ----
    gemm(zh, LAT_, Wzch, LAT_, inibuf, G_, nullptr, 0, b_zc, nullptr, nullptr, 0, B_, G_, LAT_);
    { int tot = B_ * HC_;
      k_scatter_init<<<(tot + 255) / 256, 256, 0, stream>>>(inibuf, cat_c, cc0, cc1, B_, HC_); }

    // ---- Phase 2: conductor 2-layer LSTM over NB bars (layer-0 x == 0) ----
    for (int nb = 0; nb < NB_; ++nb) {
        gemm(cat_c, 2 * HC_, Wc0h, HC_, gat_c, G_, nullptr, 0,
             c_bih0, c_bhh0, nullptr, 0, B_, G_, HC_);
        { int tot = B_ * HC_;
          k_lstm_cell<<<(tot + 255) / 256, 256, 0, stream>>>(gat_c, cc0, cat_c, 2 * HC_,
                                                             (_Float16*)nullptr, 0, B_, HC_); }
        gemm(cat_c, 2 * HC_, Wc1c, 2 * HC_, gat_c, G_, nullptr, 0,
             c_bih1, c_bhh1, nullptr, 0, B_, G_, 2 * HC_);
        { int tot = B_ * HC_;
          k_lstm_cell<<<(tot + 255) / 256, 256, 0, stream>>>(gat_c, cc1, cat_c + HC_, 2 * HC_,
                                                             condh + (size_t)nb * B_ * HC_, HC_,
                                                             B_, HC_); }
    }

    // ---- Phase 3: decoder initial states: di = cond @ cd_w^T + cd_b ----
    gemm(condh, HC_, Wcdh, HC_, gat_d, G_, nullptr, 0, cd_b, nullptr, nullptr, 0, RB_, G_, HC_);
    { int tot = RB_ * HD_;
      k_scatter_init<<<(tot + 255) / 256, 256, 0, stream>>>(gat_d, cat_d, dc0, dc1, RB_, HD_); }

    // ---- Phase 4: xs = [prev @ np_w^T + np_b  |  bar broadcast] (f16) ----
    { int tot = TR_ * NOTES_;
      k_build_prev<<<(tot + 255) / 256, 256, 0, stream>>>(tgt, prevb); }
    gemm(prevb, NOTES_, Wnph, NOTES_, nullptr, 0, xs, E_ + HC_,
         np_b, nullptr, nullptr, 0, TR_, E_, NOTES_);
    { int tot = TR_ * HC_;
      k_bcast_bar<<<(tot + 255) / 256, 256, 0, stream>>>(condh, xs); }

    // ---- Phase 5: decoder 2-layer LSTM over S steps ----
    for (int t = 0; t < S_; ++t) {
        const _Float16* xst = xs + (size_t)t * RB_ * (E_ + HC_);
        // gates0 = xs_t @ d_wih0^T + biases, then += h0 @ d_whh0^T
        gemm(xst, E_ + HC_, Wd0i, E_ + HC_, gat_d, G_, nullptr, 0,
             d_bih0, d_bhh0, nullptr, 0, RB_, G_, E_ + HC_);
        gemm(cat_d, 2 * HD_, Wd0h, HD_, gat_d, G_, nullptr, 0,
             nullptr, nullptr, gat_d, G_, RB_, G_, HD_);
        { int tot = RB_ * HD_;
          k_lstm_cell<<<(tot + 255) / 256, 256, 0, stream>>>(gat_d, dc0, cat_d, 2 * HD_,
                                                             (_Float16*)nullptr, 0, RB_, HD_); }
        // gates1 = [h0_t | h1_{t-1}] @ [d_wih1 | d_whh1]^T + biases
        gemm(cat_d, 2 * HD_, Wd1c, 2 * HD_, gat_d, G_, nullptr, 0,
             d_bih1, d_bhh1, nullptr, 0, RB_, G_, 2 * HD_);
        { int tot = RB_ * HD_;
          k_lstm_cell<<<(tot + 255) / 256, 256, 0, stream>>>(gat_d, dc1, cat_d + HD_, 2 * HD_,
                                                             ys + (size_t)t * RB_ * HD_, HD_,
                                                             RB_, HD_); }
    }

    // ---- Phase 6: logits = ys @ op_w^T + op_b ; permute to output layout ----
    gemm(ys, HD_, Woph, HD_, lgts, OUT_, nullptr, 0, op_b, nullptr, nullptr, 0, TR_, OUT_, HD_);
    { int tot = TR_ * OUT_;
      k_scatter_out<<<(tot + 255) / 256, 256, 0, stream>>>(lgts, out); }
}
